// PredModel_62345745269214
// MI455X (gfx1250) — matrix operand order repaired and checked by reading, not compiled, not run on hardware
//
#include <hip/hip_runtime.h>

// ---------------- problem constants ----------------
#define B_       64
#define T_AUDIO_ 2048
#define FEAT_    256
#define H_       1024
#define C_       96
#define E_       256
#define T_CHAR_  200

// ---------------- launch geometry ----------------
#define NBLK   64                 // persistent blocks (grid-sync'd); block b -> nt=b
#define NTHR   128                // 4 waves/block; wave w -> mt=w
#define KB_ENC 40                 // (FEAT+H)/32 k-blocks for [x|h] @ [W;U]
#define KB_FC  32                 // H/32 k-blocks for fc
#define KBG    4                  // k-blocks staged per workgroup barrier
#define NGRP   (KB_ENC / KBG)     // 10 stage groups per timestep

typedef __attribute__((ext_vector_type(16))) __bf16 v16bf;
typedef __attribute__((ext_vector_type(8)))  __bf16 v8bf;
typedef __attribute__((ext_vector_type(8)))  float  v8f;

#define WENC_ELEMS ((size_t)256 * KB_ENC * 512)        // 5,242,880 bf16 (global, swizzled)
#define WFC_ELEMS  ((size_t)6 * KB_FC * 512)           //    98,304 bf16
#define EMB_ELEMS  ((size_t)C_ * E_)
#define HBUF_ELEMS ((size_t)B_ * H_)

// ---------------- LDS carve (elements of bf16) ----------------
#define WLDS_ELEMS  ((size_t)4 * KB_ENC * 512)         // 81,920  (160 KB) gate weights
#define FCLDS_ELEMS ((size_t)KB_FC * 512)              // 16,384  ( 32 KB) fc weights
#define AP_STRIDE   40                                 // padded row stride (80 B)
#define APAN_ELEMS  ((size_t)2 * KBG * 64 * AP_STRIDE) // 20,480  ( 40 KB) A panels x2
#define SMEM_BYTES  ((WLDS_ELEMS + FCLDS_ELEMS + APAN_ELEMS) * 2)   // 237,568 B < 320 KB

__device__ __forceinline__ float sigmoidf_(float x) { return 1.0f / (1.0f + __expf(-x)); }

// ---------------------------------------------------------------------------
// Prep: swizzle weights into WMMA B-fragment order (bf16), convert embeddings,
// zero h buffers and the grid-barrier counter. B-fragment layout (32x16 bf16):
//   lane l holds column n = l%16 ; K values k = 8*(l/16) + slot (slot<8), 16 + 8*(l/16) + slot-8 (slot>=8)
// ---------------------------------------------------------------------------
__global__ void prep_kernel(const float* __restrict__ enc_W, const float* __restrict__ enc_U,
                            const float* __restrict__ dec_W, const float* __restrict__ dec_U,
                            const float* __restrict__ fc_W,  const float* __restrict__ embed,
                            __bf16* __restrict__ wenc, __bf16* __restrict__ wdec,
                            __bf16* __restrict__ wfc,  __bf16* __restrict__ emb,
                            __bf16* __restrict__ hbuf, unsigned* __restrict__ ctr)
{
    size_t idx    = (size_t)blockIdx.x * blockDim.x + threadIdx.x;
    size_t stride = (size_t)gridDim.x * blockDim.x;

    for (size_t i = idx; i < WENC_ELEMS; i += stride) {
        int slot = (int)(i & 15);
        int lane = (int)((i >> 4) & 31);
        size_t r = i >> 9;                  // ct*KB_ENC + kb
        int kb = (int)(r % KB_ENC);
        int ct = (int)(r / KB_ENC);
        int n  = ct * 16 + (lane & 15);
        int k  = kb * 32 + ((slot >> 3) * 16) + ((lane >> 4) * 8) + (slot & 7);
        float ve = (k < FEAT_) ? enc_W[(size_t)k * (4 * H_) + n]
                               : enc_U[(size_t)(k - FEAT_) * (4 * H_) + n];
        wenc[i] = (__bf16)ve;
        float vd = (k < E_) ? dec_W[(size_t)k * (4 * H_) + n]
                            : dec_U[(size_t)(k - E_) * (4 * H_) + n];
        wdec[i] = (__bf16)vd;
    }
    for (size_t i = idx; i < WFC_ELEMS; i += stride) {
        int slot = (int)(i & 15);
        int lane = (int)((i >> 4) & 31);
        size_t r = i >> 9;
        int kb = (int)(r % KB_FC);
        int ct = (int)(r / KB_FC);
        int n  = ct * 16 + (lane & 15);
        int k  = kb * 32 + ((slot >> 3) * 16) + ((lane >> 4) * 8) + (slot & 7);
        wfc[i] = (__bf16)fc_W[(size_t)k * C_ + n];
    }
    for (size_t i = idx; i < EMB_ELEMS; i += stride) emb[i] = (__bf16)embed[i];
    for (size_t i = idx; i < 2 * HBUF_ELEMS; i += stride) hbuf[i] = (__bf16)0.0f;
    if (idx < 16) ctr[idx] = 0u;
}

// ---------------------------------------------------------------------------
#define GRID_SYNC()                                                                         \
    do {                                                                                    \
        ++phase;                                                                            \
        __threadfence();                                                                    \
        __syncthreads();                                                                    \
        if (threadIdx.x == 0) {                                                             \
            __hip_atomic_fetch_add(ctr, 1u, __ATOMIC_ACQ_REL, __HIP_MEMORY_SCOPE_AGENT);    \
            while (__hip_atomic_load(ctr, __ATOMIC_ACQUIRE, __HIP_MEMORY_SCOPE_AGENT) <     \
                   phase * (unsigned)NBLK) {                                                \
                __builtin_amdgcn_s_sleep(2);                                                \
            }                                                                               \
        }                                                                                   \
        __syncthreads();                                                                    \
    } while (0)

// ---------------------------------------------------------------------------
// Persistent seq2seq kernel. Block nt keeps its gate weights resident in LDS;
// the [x_t|h] / [emb|h] A-panel is staged KBG k-blocks at a time into
// double-buffered LDS shared by all 4 waves. c-state stays in registers.
// ---------------------------------------------------------------------------
__global__ void __launch_bounds__(NTHR, 1)
lstm_seq2seq(const float* __restrict__ x,
             const float* __restrict__ enc_b, const float* __restrict__ dec_b,
             const float* __restrict__ fc_b,
             const __bf16* __restrict__ wenc, const __bf16* __restrict__ wdec,
             const __bf16* __restrict__ wfc,  const __bf16* __restrict__ emb,
             __bf16* __restrict__ hbuf0, __bf16* __restrict__ hbuf1,
             float* __restrict__ logits, int* __restrict__ tok,
             unsigned* __restrict__ ctr, const int* __restrict__ sot,
             float* __restrict__ out_preds, float* __restrict__ out_probs)
{
    extern __shared__ __bf16 smem[];
    __bf16* wlds  = smem;                               // [gate][kb][lane][16]
    __bf16* fclds = smem + WLDS_ELEMS;                  // [kb][lane][16]
    __bf16* apan  = smem + WLDS_ELEMS + FCLDS_ELEMS;    // [2][KBG][64][AP_STRIDE]

    const int tid  = threadIdx.x;
    const int lane = tid & 31;
    const int mt   = tid >> 5;          // wave id == batch row tile 0..3
    const int nt   = blockIdx.x;        // H column tile 0..63
    const int nl   = lane & 15;
    const int half = lane >> 4;
    const int ncol = nt * 16 + nl;      // output column for this lane
    const int srow = tid >> 1;          // staging row 0..63
    const int scol = (tid & 1) * 16;    // staging col base 0/16
    unsigned phase = 0;

    // ---- load this block's encoder gate weights + fc weights into LDS ----
    #pragma unroll
    for (int g = 0; g < 4; ++g) {
        const uint4* src = (const uint4*)(wenc + (size_t)(g * 64 + nt) * KB_ENC * 512);
        uint4* dst = (uint4*)(wlds + (size_t)g * KB_ENC * 512);
        for (int i = tid; i < KB_ENC * 512 / 8; i += NTHR) dst[i] = src[i];
    }
    if (nt < 6) {
        const uint4* src = (const uint4*)(wfc + (size_t)nt * KB_FC * 512);
        uint4* dst = (uint4*)fclds;
        for (int i = tid; i < KB_FC * 512 / 8; i += NTHR) dst[i] = src[i];
    }
    __syncthreads();

    // gate biases (column-indexed -> one scalar per lane)
    const float bEi = enc_b[0 * H_ + ncol], bEf = enc_b[1 * H_ + ncol];
    const float bEg = enc_b[2 * H_ + ncol], bEo = enc_b[3 * H_ + ncol];
    const float bDi = dec_b[0 * H_ + ncol], bDf = dec_b[1 * H_ + ncol];
    const float bDg = dec_b[2 * H_ + ncol], bDo = dec_b[3 * H_ + ncol];

    v8f cst = {};                        // persistent cell-state tile
    __bf16* hb[2] = {hbuf0, hbuf1};
    int cur = 0;

    const int arow = mt * 16 + nl;       // A-fragment row within the panel

    // =========================== encoder ===========================
    for (int t = 0; t < T_AUDIO_; ++t) {
        const __bf16* hcur = hb[cur];
        v8f acc[4] = {v8f{}, v8f{}, v8f{}, v8f{}};

        for (int grp = 0; grp < NGRP; ++grp) {
            const int buf = grp & 1;
            __bf16* pan = apan + (size_t)buf * KBG * 64 * AP_STRIDE;

            // ---- cooperative stage of KBG 64x32 A panels ----
            #pragma unroll
            for (int kk = 0; kk < KBG; ++kk) {
                const int kbase = (grp * KBG + kk) * 32;
                __bf16* dst = pan + ((size_t)kk * 64 + srow) * AP_STRIDE + scol;
                if (kbase < FEAT_) {                    // x_t slice (f32 -> bf16)
                    const float* xp = x + ((size_t)srow * T_AUDIO_ + t) * FEAT_ + kbase + scol;
                    v8f f0 = *(const v8f*)xp;
                    v8f f1 = *(const v8f*)(xp + 8);
                    if (kbase == 0) __builtin_prefetch(xp + FEAT_, 0, 1);   // next timestep
                    #pragma unroll
                    for (int j = 0; j < 8; ++j) { dst[j] = (__bf16)f0[j]; dst[8 + j] = (__bf16)f1[j]; }
                } else {                                // h slice (bf16)
                    v16bf hv = *(const v16bf*)(hcur + (size_t)srow * H_ + (kbase - FEAT_) + scol);
                    #pragma unroll
                    for (int j = 0; j < 16; ++j) dst[j] = hv[j];
                }
            }
            __syncthreads();

            // ---- KBG k-steps: A from LDS panel, B from resident LDS weights ----
            #pragma unroll 2
            for (int kk = 0; kk < KBG; ++kk) {
                const int kb = grp * KBG + kk;
                const __bf16* ap = pan + ((size_t)kk * 64 + arow) * AP_STRIDE;
                v8bf a0 = *(const v8bf*)(ap + half * 8);
                v8bf a1 = *(const v8bf*)(ap + 16 + half * 8);
                v16bf a;
                #pragma unroll
                for (int j = 0; j < 8; ++j) { a[j] = a0[j]; a[8 + j] = a1[j]; }
                const __bf16* wb = wlds + ((size_t)kb * 32 + lane) * 16;
                v16bf b0 = *(const v16bf*)(wb + (size_t)0 * KB_ENC * 512);
                v16bf b1 = *(const v16bf*)(wb + (size_t)1 * KB_ENC * 512);
                v16bf b2 = *(const v16bf*)(wb + (size_t)2 * KB_ENC * 512);
                v16bf b3 = *(const v16bf*)(wb + (size_t)3 * KB_ENC * 512);
                acc[0] = __builtin_amdgcn_wmma_f32_16x16x32_bf16(false, a, false, b0, (short)0, acc[0], false, false);
                acc[1] = __builtin_amdgcn_wmma_f32_16x16x32_bf16(false, a, false, b1, (short)0, acc[1], false, false);
                acc[2] = __builtin_amdgcn_wmma_f32_16x16x32_bf16(false, a, false, b2, (short)0, acc[2], false, false);
                acc[3] = __builtin_amdgcn_wmma_f32_16x16x32_bf16(false, a, false, b3, (short)0, acc[3], false, false);
            }
        }
        __bf16* hout = hb[cur ^ 1] + (size_t)(mt * 16 + half * 8) * H_ + ncol;
        #pragma unroll
        for (int v = 0; v < 8; ++v) {
            float ig = sigmoidf_(acc[0][v] + bEi);
            float fg = sigmoidf_(acc[1][v] + bEf);
            float gg = tanhf(acc[2][v] + bEg);
            float og = sigmoidf_(acc[3][v] + bEo);
            float cn = fg * cst[v] + ig * gg;
            cst[v] = cn;
            hout[(size_t)v * H_] = (__bf16)(og * tanhf(cn));
        }
        cur ^= 1;
        GRID_SYNC();
    }

    // ---- swap in decoder weights; seed tokens ----
    #pragma unroll
    for (int g = 0; g < 4; ++g) {
        const uint4* src = (const uint4*)(wdec + (size_t)(g * 64 + nt) * KB_ENC * 512);
        uint4* dst = (uint4*)(wlds + (size_t)g * KB_ENC * 512);
        for (int i = tid; i < KB_ENC * 512 / 8; i += NTHR) dst[i] = src[i];
    }
    if (blockIdx.x == 0 && tid < B_) tok[tid] = *sot;
    __syncthreads();
    GRID_SYNC();

    // =========================== decoder ===========================
    for (int s = 0; s < T_CHAR_; ++s) {
        const __bf16* hcur = hb[cur];
        const int myTok = tok[srow];                   // embedding row for staging
        v8f acc[4] = {v8f{}, v8f{}, v8f{}, v8f{}};

        for (int grp = 0; grp < NGRP; ++grp) {
            const int buf = grp & 1;
            __bf16* pan = apan + (size_t)buf * KBG * 64 * AP_STRIDE;

            #pragma unroll
            for (int kk = 0; kk < KBG; ++kk) {
                const int kbase = (grp * KBG + kk) * 32;
                __bf16* dst = pan + ((size_t)kk * 64 + srow) * AP_STRIDE + scol;
                const __bf16* srcp = (kbase < E_)
                    ? (emb + (size_t)myTok * E_ + kbase + scol)
                    : (hcur + (size_t)srow * H_ + (kbase - E_) + scol);
                v16bf hv = *(const v16bf*)srcp;
                #pragma unroll
                for (int j = 0; j < 16; ++j) dst[j] = hv[j];
            }
            __syncthreads();

            #pragma unroll 2
            for (int kk = 0; kk < KBG; ++kk) {
                const int kb = grp * KBG + kk;
                const __bf16* ap = pan + ((size_t)kk * 64 + arow) * AP_STRIDE;
                v8bf a0 = *(const v8bf*)(ap + half * 8);
                v8bf a1 = *(const v8bf*)(ap + 16 + half * 8);
                v16bf a;
                #pragma unroll
                for (int j = 0; j < 8; ++j) { a[j] = a0[j]; a[8 + j] = a1[j]; }
                const __bf16* wb = wlds + ((size_t)kb * 32 + lane) * 16;
                v16bf b0 = *(const v16bf*)(wb + (size_t)0 * KB_ENC * 512);
                v16bf b1 = *(const v16bf*)(wb + (size_t)1 * KB_ENC * 512);
                v16bf b2 = *(const v16bf*)(wb + (size_t)2 * KB_ENC * 512);
                v16bf b3 = *(const v16bf*)(wb + (size_t)3 * KB_ENC * 512);
                acc[0] = __builtin_amdgcn_wmma_f32_16x16x32_bf16(false, a, false, b0, (short)0, acc[0], false, false);
                acc[1] = __builtin_amdgcn_wmma_f32_16x16x32_bf16(false, a, false, b1, (short)0, acc[1], false, false);
                acc[2] = __builtin_amdgcn_wmma_f32_16x16x32_bf16(false, a, false, b2, (short)0, acc[2], false, false);
                acc[3] = __builtin_amdgcn_wmma_f32_16x16x32_bf16(false, a, false, b3, (short)0, acc[3], false, false);
            }
        }
        __bf16* hout = hb[cur ^ 1] + (size_t)(mt * 16 + half * 8) * H_ + ncol;
        #pragma unroll
        for (int v = 0; v < 8; ++v) {
            float ig = sigmoidf_(acc[0][v] + bDi);
            float fg = sigmoidf_(acc[1][v] + bDf);
            float gg = tanhf(acc[2][v] + bDg);
            float og = sigmoidf_(acc[3][v] + bDo);
            float cn = fg * cst[v] + ig * gg;
            cst[v] = cn;
            hout[(size_t)v * H_] = (__bf16)(og * tanhf(cn));
        }
        cur ^= 1;
        GRID_SYNC();

        // ---- logits = h @ fc_W + fc_b (blocks nt<6; fc weights LDS-resident) ----
        if (nt < 6) {
            const __bf16* hr2 = hb[cur] + (size_t)(mt * 16 + nl) * H_;
            v8f la = {};
            for (int kb = 0; kb < KB_FC; ++kb) {
                const __bf16* hp = hr2 + kb * 32;
                v8bf h0 = *(const v8bf*)(hp + half * 8);
                v8bf h1 = *(const v8bf*)(hp + 16 + half * 8);
                v16bf a;
                #pragma unroll
                for (int j = 0; j < 8; ++j) { a[j] = h0[j]; a[8 + j] = h1[j]; }
                const v16bf bm = *(const v16bf*)(fclds + ((size_t)kb * 32 + lane) * 16);
                la = __builtin_amdgcn_wmma_f32_16x16x32_bf16(
                         false, a, false, bm, (short)0, la, false, false);
            }
            const float bfc = fc_b[ncol];
            #pragma unroll
            for (int v = 0; v < 8; ++v)
                logits[(size_t)(mt * 16 + half * 8 + v) * C_ + nt * 16 + nl] = la[v] + bfc;
        }
        GRID_SYNC();

        // ---- softmax + argmax, emit preds/probs, feed token back ----
        if (blockIdx.x == 0 && tid < B_) {
            const int r = tid;
            const float* lr = logits + (size_t)r * C_;
            float mx = lr[0];
            int arg = 0;
            for (int j = 1; j < C_; ++j) { float v = lr[j]; if (v > mx) { mx = v; arg = j; } }
            float sum = 0.0f;
            for (int j = 0; j < C_; ++j) sum += __expf(lr[j] - mx);
            const float inv = 1.0f / sum;
            float* pp = out_probs + ((size_t)s * B_ + r) * C_;
            for (int j = 0; j < C_; ++j) pp[j] = __expf(lr[j] - mx) * inv;
            out_preds[(size_t)s * B_ + r] = (float)arg;
            tok[r] = arg;
        }
        GRID_SYNC();
    }
}

// ---------------------------------------------------------------------------
extern "C" void kernel_launch(void* const* d_in, const int* in_sizes, int n_in,
                              void* d_out, int out_size, void* d_ws, size_t ws_size,
                              hipStream_t stream) {
    (void)in_sizes; (void)n_in; (void)out_size; (void)ws_size;

    const float* x     = (const float*)d_in[0];
    const float* enc_W = (const float*)d_in[1];
    const float* enc_U = (const float*)d_in[2];
    const float* enc_b = (const float*)d_in[3];
    const float* embed = (const float*)d_in[4];
    const float* dec_W = (const float*)d_in[5];
    const float* dec_U = (const float*)d_in[6];
    const float* dec_b = (const float*)d_in[7];
    const float* fc_W  = (const float*)d_in[8];
    const float* fc_b  = (const float*)d_in[9];
    const int*   sot   = (const int*)d_in[10];

    char* ws = (char*)d_ws;
    size_t off = 0;
    auto alloc = [&](size_t bytes) -> void* {
        void* p = ws + off;
        off += (bytes + 255) & ~(size_t)255;
        return p;
    };

    unsigned* ctr    = (unsigned*)alloc(64);
    int*      tok    = (int*)alloc(B_ * sizeof(int));
    float*    logits = (float*)alloc((size_t)B_ * C_ * sizeof(float));
    __bf16*   hbuf0  = (__bf16*)alloc(HBUF_ELEMS * sizeof(__bf16));
    __bf16*   hbuf1  = (__bf16*)alloc(HBUF_ELEMS * sizeof(__bf16));
    __bf16*   wenc   = (__bf16*)alloc(WENC_ELEMS * sizeof(__bf16));
    __bf16*   wdec   = (__bf16*)alloc(WENC_ELEMS * sizeof(__bf16));
    __bf16*   wfc    = (__bf16*)alloc(WFC_ELEMS * sizeof(__bf16));
    __bf16*   emb    = (__bf16*)alloc(EMB_ELEMS * sizeof(__bf16));

    float* out_preds = (float*)d_out;                       // [T_CHAR, B] (as float)
    float* out_probs = out_preds + (size_t)T_CHAR_ * B_;    // [T_CHAR, B, C]

    // allow >64KB dynamic LDS (idempotent, capture-safe host call)
    (void)hipFuncSetAttribute((const void*)lstm_seq2seq,
                              hipFuncAttributeMaxDynamicSharedMemorySize,
                              (int)SMEM_BYTES);

    prep_kernel<<<2048, 256, 0, stream>>>(enc_W, enc_U, dec_W, dec_U, fc_W, embed,
                                          wenc, wdec, wfc, emb, hbuf0, ctr);

    lstm_seq2seq<<<NBLK, NTHR, SMEM_BYTES, stream>>>(x, enc_b, dec_b, fc_b,
                                                     wenc, wdec, wfc, emb,
                                                     hbuf0, hbuf1, logits, tok, ctr, sot,
                                                     out_preds, out_probs);
}
